// EEGGraphConvNetLSTM_27049704030291
// MI455X (gfx1250) — matrix operand order repaired and checked by reading, not compiled, not run on hardware
//
#include <hip/hip_runtime.h>

// ---------------------------------------------------------------------------
// Types for CDNA5 WMMA (wave32): D(16x16 f32) = A(16x32 bf16) x B(32x16 bf16) + C
// ---------------------------------------------------------------------------
typedef __bf16 v16bf __attribute__((ext_vector_type(16)));
typedef float  v8f   __attribute__((ext_vector_type(8)));

union Frag16 {            // 32 bytes = 8 VGPRs
    v16bf          v;
    unsigned short us[16];
    uint4          q[2];
};

__device__ __forceinline__ unsigned short f32_to_bf16(float f) {
    unsigned int u = __float_as_uint(f);
    u += 0x7fffu + ((u >> 16) & 1u);   // round-to-nearest-even
    return (unsigned short)(u >> 16);
}

__device__ __forceinline__ float leaky(float v) { return v >= 0.f ? v : 0.01f * v; }

// ---------------------------------------------------------------------------
// Utility kernels
// ---------------------------------------------------------------------------
__global__ void zero_f32(float* p, int n) {
    int i = blockIdx.x * blockDim.x + threadIdx.x;
    if (i < n) p[i] = 0.f;
}

__global__ void cvt_f32_bf16(const float* __restrict__ src, unsigned short* __restrict__ dst, int n) {
    int i = blockIdx.x * blockDim.x + threadIdx.x;
    if (i < n) dst[i] = f32_to_bf16(src[i]);
}

__global__ void add_bias2(const float* __restrict__ a, const float* __restrict__ b,
                          float* __restrict__ o, int n) {
    int i = blockIdx.x * blockDim.x + threadIdx.x;
    if (i < n) o[i] = a[i] + b[i];
}

// Build per-graph dense weighted adjacency: adj[g][dst_local][src_local] += ew
__global__ void build_adj(const int* __restrict__ src, const int* __restrict__ dst,
                          const float* __restrict__ ew, float* __restrict__ adj, int nE) {
    int e = blockIdx.x * blockDim.x + threadIdx.x;
    if (e >= nE) return;
    int s = src[e], d = dst[e];
    int g = d >> 6;
    atomicAdd(&adj[(size_t)g * 4096 + (size_t)(d & 63) * 64 + (s & 63)], ew[e]);
}

// ---------------------------------------------------------------------------
// Main GEMM: C[N,M] = A[N,K](bf16) * W[M,K]^T(bf16)   (one wave per 16x16 tile)
// Optionally adds per-column bias; stores bf16 or f32.
// ---------------------------------------------------------------------------
__global__ void gemm_wmma(const unsigned short* __restrict__ A,
                          const unsigned short* __restrict__ W,
                          void* __restrict__ Cout,
                          const float* __restrict__ bias,
                          int N, int K, int M, int storeBf16) {
    int lane = threadIdx.x & 31;
    int wave = threadIdx.x >> 5;
    int tm = blockIdx.x;                  // column tile (M/16)
    int tn = blockIdx.y * 4 + wave;       // row tile (N/16)
    int half = lane >> 4;
    int r16  = lane & 15;

    const unsigned short* aBase = A + (size_t)(tn * 16 + r16) * K;
    const unsigned short* wBase = W + (size_t)(tm * 16 + r16) * K;

    v8f acc = {};
    for (int k0 = 0; k0 < K; k0 += 32) {
        Frag16 a, b;
        a.q[0] = *(const uint4*)(aBase + k0 + half * 8);
        a.q[1] = *(const uint4*)(aBase + k0 + 16 + half * 8);
        b.q[0] = *(const uint4*)(wBase + k0 + half * 8);
        b.q[1] = *(const uint4*)(wBase + k0 + 16 + half * 8);
        acc = __builtin_amdgcn_wmma_f32_16x16x32_bf16(
            false, a.v, false, b.v, (short)0, acc, false, false);
    }

    int col = tm * 16 + r16;
    float bv = bias ? bias[col] : 0.f;
    if (storeBf16) {
        unsigned short* C = (unsigned short*)Cout;
#pragma unroll
        for (int e = 0; e < 8; ++e) {
            int row = tn * 16 + half * 8 + e;
            C[(size_t)row * M + col] = f32_to_bf16(acc[e] + bv);
        }
    } else {
        float* C = (float*)Cout;
#pragma unroll
        for (int e = 0; e < 8; ++e) {
            int row = tn * 16 + half * 8 + e;
            C[(size_t)row * M + col] = acc[e] + bv;
        }
    }
}

// ---------------------------------------------------------------------------
// Scatter as batched dense GEMM: Z_g[64,F] = Adj_g[64,64] * Y_g[64,F] + bias
// grid = (F/16, 4, 128), one wave per block.
// ---------------------------------------------------------------------------
__global__ void scatter_gemm_wmma(const unsigned short* __restrict__ adjB,
                                  const unsigned short* __restrict__ Yb,
                                  float* __restrict__ Z,
                                  const float* __restrict__ bias, int F) {
    int lane = threadIdx.x & 31;
    int g  = blockIdx.z;
    int tr = blockIdx.y;      // dst row tile (0..3)
    int tm = blockIdx.x;      // feature tile
    int half = lane >> 4;
    int r16  = lane & 15;

    const unsigned short* aBase = adjB + (size_t)g * 4096 + (size_t)(tr * 16 + r16) * 64;
    int col = tm * 16 + r16;
    const unsigned short* yBase = Yb + (size_t)g * 64 * F + col;

    v8f acc = {};
#pragma unroll
    for (int k0 = 0; k0 < 64; k0 += 32) {
        Frag16 a, b;
        a.q[0] = *(const uint4*)(aBase + k0 + half * 8);
        a.q[1] = *(const uint4*)(aBase + k0 + 16 + half * 8);
#pragma unroll
        for (int e = 0; e < 16; ++e)
            b.us[e] = yBase[(size_t)(k0 + half * 8 + (e & 7) + ((e >> 3) << 4)) * F];
        acc = __builtin_amdgcn_wmma_f32_16x16x32_bf16(
            false, a.v, false, b.v, (short)0, acc, false, false);
    }
    float bv = bias[col];
#pragma unroll
    for (int e = 0; e < 8; ++e) {
        int row = g * 64 + tr * 16 + half * 8 + e;
        Z[(size_t)row * F + col] = acc[e] + bv;
    }
}

// ---------------------------------------------------------------------------
// BatchNorm (training stats) + leaky, output bf16 for next WMMA stage
// ---------------------------------------------------------------------------
__global__ void bn_stats(const float* __restrict__ Z, float* __restrict__ sums,
                         int N, int F) {
    int row0 = blockIdx.x * 64;
    for (int c = threadIdx.x; c < F; c += blockDim.x) {
        float s = 0.f, sq = 0.f;
        for (int r = 0; r < 64; ++r) {
            float v = Z[(size_t)(row0 + r) * F + c];
            s += v; sq += v * v;
        }
        atomicAdd(&sums[c], s);
        atomicAdd(&sums[F + c], sq);
    }
}

__global__ void bn_finalize(const float* __restrict__ sums, float* __restrict__ mv,
                            int N, int F) {
    int c = blockIdx.x * blockDim.x + threadIdx.x;
    if (c >= F) return;
    float mean = sums[c] / (float)N;
    float var  = sums[F + c] / (float)N - mean * mean;
    mv[c]     = mean;
    mv[F + c] = rsqrtf(var + 1e-5f);
}

__global__ void bn_apply_leaky_bf16(const float* __restrict__ Z,
                                    const float* __restrict__ mv,
                                    const float* __restrict__ gam,
                                    const float* __restrict__ bet,
                                    unsigned short* __restrict__ out,
                                    int N, int F) {
    int i = blockIdx.x * blockDim.x + threadIdx.x;
    if (i >= N * F) return;
    int c = i % F;
    float y = gam[c] * (Z[i] - mv[c]) * mv[F + c] + bet[c];
    out[i] = f32_to_bf16(leaky(y));
}

// ---------------------------------------------------------------------------
// Sequential LSTM scan: single persistent workgroup (1024 threads).
// Whh (1MB f32) stays resident in the 192MB L2; h/gates in LDS.
// Pool partial sums accumulate in registers, flushed every 64 steps.
// ---------------------------------------------------------------------------
__global__ void lstm_scan(const float* __restrict__ pre,   // [8192,1024]
                          const float* __restrict__ Whh,   // [1024,256]
                          float* __restrict__ pool) {      // [128,256]
    __shared__ __align__(16) float hsh[256];
    __shared__ __align__(16) float gsh[1024];
    int j = threadIdx.x;
    float c = 0.f, pacc = 0.f;
    if (j < 256) hsh[j] = 0.f;
    __syncthreads();
    const float4* wrow = (const float4*)(Whh + (size_t)j * 256);
    for (int t = 0; t < 8192; ++t) {
        if (t + 1 < 8192)
            __builtin_prefetch(&pre[(size_t)(t + 1) * 1024 + j], 0, 0);
        float acc = pre[(size_t)t * 1024 + j];
        const float4* hv = (const float4*)hsh;
#pragma unroll 8
        for (int k = 0; k < 64; ++k) {
            float4 w = wrow[k];
            float4 h = hv[k];
            acc += w.x * h.x + w.y * h.y + w.z * h.z + w.w * h.w;
        }
        gsh[j] = acc;
        __syncthreads();
        if (j < 256) {
            float ig = 1.f / (1.f + __expf(-gsh[j]));
            float fg = 1.f / (1.f + __expf(-gsh[256 + j]));
            float gg = tanhf(gsh[512 + j]);
            float og = 1.f / (1.f + __expf(-gsh[768 + j]));
            c = fg * c + ig * gg;
            float hn = og * tanhf(c);
            hsh[j] = hn;
            pacc += hn;
            if ((t & 63) == 63) { pool[(size_t)(t >> 6) * 256 + j] = pacc; pacc = 0.f; }
        }
        __syncthreads();
    }
}

// ---------------------------------------------------------------------------
// Head MLP: 256 -> 128 -> 64 -> 2, leaky everywhere. One block per graph.
// ---------------------------------------------------------------------------
__global__ void mlp_head(const float* __restrict__ pool,
                         const float* __restrict__ fW1, const float* __restrict__ fb1,
                         const float* __restrict__ fW2, const float* __restrict__ fb2,
                         const float* __restrict__ fW3, const float* __restrict__ fb3,
                         float* __restrict__ out) {
    __shared__ float p[256], l1[128], l2[64];
    int g = blockIdx.x, t = threadIdx.x;   // 128 threads
    p[t]       = pool[(size_t)g * 256 + t];
    p[128 + t] = pool[(size_t)g * 256 + 128 + t];
    __syncthreads();
    float v = fb1[t];
    for (int k = 0; k < 256; ++k) v += fW1[(size_t)t * 256 + k] * p[k];
    l1[t] = leaky(v);
    __syncthreads();
    if (t < 64) {
        float v2 = fb2[t];
        for (int k = 0; k < 128; ++k) v2 += fW2[(size_t)t * 128 + k] * l1[k];
        l2[t] = leaky(v2);
    }
    __syncthreads();
    if (t < 2) {
        float v3 = fb3[t];
        for (int k = 0; k < 64; ++k) v3 += fW3[(size_t)t * 64 + k] * l2[k];
        out[(size_t)g * 2 + t] = leaky(v3);
    }
}

// ---------------------------------------------------------------------------
// Host launcher
// ---------------------------------------------------------------------------
extern "C" void kernel_launch(void* const* d_in, const int* in_sizes, int n_in,
                              void* d_out, int out_size, void* d_ws, size_t ws_size,
                              hipStream_t stream) {
    constexpr int N = 8192, G = 128, E = 131072;
    const float* x   = (const float*)d_in[0];
    const int*   ei  = (const int*)d_in[1];
    const float* ew  = (const float*)d_in[2];
    const float* W1  = (const float*)d_in[4];
    const float* b1  = (const float*)d_in[5];
    const float* g1  = (const float*)d_in[6];
    const float* be1 = (const float*)d_in[7];
    const float* W2  = (const float*)d_in[8];
    const float* b2  = (const float*)d_in[9];
    const float* g2  = (const float*)d_in[10];
    const float* be2 = (const float*)d_in[11];
    const float* W3  = (const float*)d_in[12];
    const float* b3  = (const float*)d_in[13];
    const float* g3  = (const float*)d_in[14];
    const float* be3 = (const float*)d_in[15];
    const float* Wih = (const float*)d_in[16];
    const float* Whh = (const float*)d_in[17];
    const float* bih = (const float*)d_in[18];
    const float* bhh = (const float*)d_in[19];
    const float* fW1 = (const float*)d_in[20];
    const float* fb1 = (const float*)d_in[21];
    const float* fW2 = (const float*)d_in[22];
    const float* fb2 = (const float*)d_in[23];
    const float* fW3 = (const float*)d_in[24];
    const float* fb3 = (const float*)d_in[25];
    float* out = (float*)d_out;

    // workspace layout
    char* ws = (char*)d_ws;
    size_t off = 0;
    auto alloc = [&](size_t bytes) {
        size_t o = off;
        off = (off + bytes + 255) & ~(size_t)255;
        return (void*)(ws + o);
    };
    unsigned short* XB   = (unsigned short*)alloc((size_t)N * 1280 * 2);
    unsigned short* W1B  = (unsigned short*)alloc((size_t)640 * 1280 * 2);
    unsigned short* W2B  = (unsigned short*)alloc((size_t)512 * 640 * 2);
    unsigned short* W3B  = (unsigned short*)alloc((size_t)256 * 512 * 2);
    unsigned short* WIHB = (unsigned short*)alloc((size_t)1024 * 256 * 2);
    float*          ADJF = (float*)alloc((size_t)G * 4096 * 4);
    unsigned short* ADJB = (unsigned short*)alloc((size_t)G * 4096 * 2);
    unsigned short* YB   = (unsigned short*)alloc((size_t)N * 640 * 2);
    float*          Z    = (float*)alloc((size_t)N * 640 * 4);
    unsigned short* HB   = (unsigned short*)alloc((size_t)N * 640 * 2);
    float*          SUMS = (float*)alloc((size_t)2 * 640 * 4);
    float*          MV   = (float*)alloc((size_t)2 * 640 * 4);
    float*          BSUM = (float*)alloc((size_t)1024 * 4);
    float*          PRE  = (float*)alloc((size_t)N * 1024 * 4);
    float*          POOL = (float*)alloc((size_t)G * 256 * 4);
    (void)ws_size; (void)n_in; (void)in_sizes; (void)out_size;

    auto blocks = [](int n, int b) { return (n + b - 1) / b; };

    // ---- precompute: bf16 copies, adjacency, combined LSTM bias ----
    cvt_f32_bf16<<<blocks(N * 1280, 256), 256, 0, stream>>>(x, XB, N * 1280);
    cvt_f32_bf16<<<blocks(640 * 1280, 256), 256, 0, stream>>>(W1, W1B, 640 * 1280);
    cvt_f32_bf16<<<blocks(512 * 640, 256), 256, 0, stream>>>(W2, W2B, 512 * 640);
    cvt_f32_bf16<<<blocks(256 * 512, 256), 256, 0, stream>>>(W3, W3B, 256 * 512);
    cvt_f32_bf16<<<blocks(1024 * 256, 256), 256, 0, stream>>>(Wih, WIHB, 1024 * 256);
    add_bias2<<<blocks(1024, 256), 256, 0, stream>>>(bih, bhh, BSUM, 1024);

    zero_f32<<<blocks(G * 4096, 256), 256, 0, stream>>>(ADJF, G * 4096);
    build_adj<<<blocks(E, 256), 256, 0, stream>>>(ei, ei + E, ew, ADJF, E);
    cvt_f32_bf16<<<blocks(G * 4096, 256), 256, 0, stream>>>(ADJF, ADJB, G * 4096);

    // ---- three GCN layers: WMMA gemm -> adjacency WMMA gemm -> BN+leaky ----
    const unsigned short* Ain = XB;
    const unsigned short* Wb[3]  = { W1B, W2B, W3B };
    const float* bb[3]  = { b1, b2, b3 };
    const float* gg[3]  = { g1, g2, g3 };
    const float* bee[3] = { be1, be2, be3 };
    int Kdim[3] = { 1280, 640, 512 };
    int Fdim[3] = { 640, 512, 256 };

    for (int L = 0; L < 3; ++L) {
        int K = Kdim[L], F = Fdim[L];
        // lin: YB[N,F] = Ain[N,K] @ W^T  (bf16 out, no bias yet)
        gemm_wmma<<<dim3(F / 16, N / 64), 128, 0, stream>>>(
            Ain, Wb[L], (void*)YB, nullptr, N, K, F, 1);
        // message passing as batched dense GEMM + bias
        scatter_gemm_wmma<<<dim3(F / 16, 4, G), 32, 0, stream>>>(ADJB, YB, Z, bb[L], F);
        // batchnorm (training stats) + leaky, bf16 out for next layer
        zero_f32<<<blocks(2 * F, 256), 256, 0, stream>>>(SUMS, 2 * F);
        bn_stats<<<N / 64, 256, 0, stream>>>(Z, SUMS, N, F);
        bn_finalize<<<blocks(F, 256), 256, 0, stream>>>(SUMS, MV, N, F);
        bn_apply_leaky_bf16<<<blocks(N * F, 256), 256, 0, stream>>>(
            Z, MV, gg[L], bee[L], HB, N, F);
        Ain = HB;
    }

    // ---- LSTM input projection: PRE[N,1024] = HB[N,256] @ Wih^T + (bih+bhh) ----
    gemm_wmma<<<dim3(1024 / 16, N / 64), 128, 0, stream>>>(
        HB, WIHB, (void*)PRE, BSUM, N, 256, 1024, 0);

    // ---- sequential scan (persistent workgroup) + fused graph pooling ----
    lstm_scan<<<1, 1024, 0, stream>>>(PRE, Whh, POOL);

    // ---- head MLP ----
    mlp_head<<<G, 128, 0, stream>>>(POOL, fW1, fb1, fW2, fb2, fW3, fb3, out);
}
